// EdgeGraphConvBlock_12498354831402
// MI455X (gfx1250) — matrix operand order repaired and checked by reading, not compiled, not run on hardware
//
#include <hip/hip_runtime.h>
#include <hip/hip_bf16.h>
#include <stdint.h>

// ---------------------------------------------------------------------------
// EdgeGraphConv block for MI455X (gfx1250, wave32, WMMA f16 16x16x32)
// B=2 N=8192 C=24 H=O=64 K=32
// ---------------------------------------------------------------------------

typedef __attribute__((ext_vector_type(16))) _Float16 v16h;
typedef __attribute__((ext_vector_type(8)))  float    v8f;

#define B_   2
#define N_   8192
#define C_   24
#define H_   64
#define O_   64
#define KNN_ 32
#define M_TOT (B_ * N_ * KNN_)      // 524288 rows of the implicit edge matrix
#define SLOPE_ 0.2f
#define EPS_   1e-5f

union Frag16 { v16h h; uint32_t u[8]; };

static __device__ __forceinline__ uint32_t pkh(float lo, float hi) {
  _Float16 a = (_Float16)lo, b = (_Float16)hi;
  uint16_t ua = __builtin_bit_cast(uint16_t, a);
  uint16_t ub = __builtin_bit_cast(uint16_t, b);
  return (uint32_t)ua | ((uint32_t)ub << 16);
}

static __device__ __forceinline__ v8f wmma16(const Frag16& A, const Frag16& Bf, v8f C) {
  // 8 args: (neg_a, A, neg_b, B, c_mod, C, reuse_a, reuse_b)
  return __builtin_amdgcn_wmma_f32_16x16x32_f16(false, A.h, false, Bf.h,
                                                (short)0, C, false, false);
}

// ---------------------------------------------------------------------------
// Workspace layout (floats):
//   [  0.. 63] sum1   [ 64..127] sumsq1
//   [128..191] sum2   [192..255] sumsq2
//   [256..319] a1     [320..383] b1      (hn = h*a + b)
//   [384..447] a2     [448..511] b2
// ---------------------------------------------------------------------------

__global__ void init_ws_kernel(float* __restrict__ ws) {
  int t = threadIdx.x;
  if (t < 256) ws[t] = 0.0f;
}

// ---------------------------------------------------------------------------
// KNN: one wave per query point. Whole-batch positions staged in LDS (96 KB,
// CDNA5 WGP has 320 KB). Exact top-33 (incl. self) via wave-parallel sorted
// insertion: ballot -> popcount -> 32-lane shift, O(1) per accepted candidate.
// Keys packed as (f32 dist bits << 32) | point id; dist >= 0 so uint order
// matches float order.
// ---------------------------------------------------------------------------
__global__ __launch_bounds__(256)
void knn_kernel(const float* __restrict__ pos, int* __restrict__ idxout) {
  extern __shared__ char smem[];
  float*    px    = (float*)smem;
  float*    py    = px + N_;
  float*    pz    = py + N_;
  uint64_t* lists = (uint64_t*)(pz + N_);   // 8 waves x 33 entries

  const int tid   = threadIdx.x;
  const int wg    = blockIdx.x;             // 2048 WGs, 8 queries each
  const int qbase = wg * 8;
  const int b     = qbase >> 13;            // batch (8192 queries per batch)

  const float* pb = pos + (size_t)b * N_ * 3;
  for (int i = tid; i < N_; i += 256) {
    px[i] = pb[i * 3 + 0];
    py[i] = pb[i * 3 + 1];
    pz[i] = pb[i * 3 + 2];
  }
  __syncthreads();

  const int wave = tid >> 5, lane = tid & 31;
  const int q    = (qbase + wave) & (N_ - 1);
  uint64_t* list = lists + wave * 33;

  list[lane] = ~0ull;
  if (lane == 0) list[32] = ~0ull;

  const float qx = px[q], qy = py[q], qz = pz[q];

  for (int j0 = 0; j0 < N_; j0 += 32) {
    const int pid = j0 + lane;
    const float dx = qx - px[pid];
    const float dy = qy - py[pid];
    const float dz = qz - pz[pid];
    const float d  = dx * dx + dy * dy + dz * dz;
    const uint64_t key =
        ((uint64_t)__builtin_bit_cast(uint32_t, d) << 32) | (uint32_t)pid;

    uint64_t worst = list[32];
    unsigned int mask = (unsigned int)__ballot(key < worst);
    while (mask) {
      const int src = __builtin_ctz(mask);
      mask &= mask - 1;
      const uint32_t khi = __shfl((uint32_t)(key >> 32), src);
      const uint32_t klo = __shfl((uint32_t)(key & 0xffffffffu), src);
      const uint64_t kb  = ((uint64_t)khi << 32) | klo;
      worst = list[32];
      if (kb >= worst) continue;            // uniform: kb/worst wave-uniform
      const uint64_t cur = list[lane];      // entries 0..31 (entry 32 = worst)
      const unsigned int lt = (unsigned int)__ballot(cur < kb);
      const int p = __popc(lt);             // insertion position, 0..32
      if (lane >= p) list[lane + 1] = cur;  // shift up; old entry 32 dropped
      if (lane == 0) list[p] = kb;
    }
  }

  // entry 0 is self (dist 0); emit entries 1..32
  const int qid = qbase + wave;
  idxout[(size_t)qid * KNN_ + lane] =
      (int)(uint32_t)(list[lane + 1] & 0xffffffffu);
}

// ---------------------------------------------------------------------------
// Fused edge-gather + GEMM1 + BN1/LReLU + GEMM2 + BN2/LReLU + k-max.
// MODE 0: accumulate BN1 stats (stop after GEMM1)
// MODE 1: apply BN1, accumulate BN2 stats (stop after GEMM2)
// MODE 2: apply both BNs, max over k, write out
// WG = 128 threads = 4 waves; 128 rows per WG = 4 points (k fastest), one
// point (32 rows = 2 WMMA row tiles) per wave -> k-max stays inside a wave.
// ---------------------------------------------------------------------------
template <int MODE>
__global__ __launch_bounds__(128)
void fused_kernel(const float* __restrict__ x,
                  const int*   __restrict__ idxbuf,
                  const float* __restrict__ w1,
                  const float* __restrict__ w2,
                  float*       __restrict__ ws,
                  float*       __restrict__ out) {
  __shared__ _Float16 edgeT[128][56];        // 128 rows x 48 f16 (+pad)
  __shared__ uint32_t B1f[4][2][32][8];      // [coltile][ktile][lane][dword]
  __shared__ uint32_t B2f[4][2][32][8];
  __shared__ _Float16 h1T[4][16][72];        // per-wave h1 tile (row-major)
  __shared__ float    ssum[64], ssq[64];

  const int tid  = threadIdx.x;
  const int wg   = blockIdx.x;
  const int lane = tid & 31;
  const int w    = tid >> 5;

  if (MODE < 2 && tid < 64) { ssum[tid] = 0.0f; ssq[tid] = 0.0f; }

  // ---- edge tile: one thread per row -------------------------------------
  {
    const int m   = wg * 128 + tid;
    const int b   = m >> 18;                 // N_*KNN_ = 2^18
    const int rem = m & ((1 << 18) - 1);
    const int n   = rem >> 5;                // KNN_ = 32
    const int id  = idxbuf[m];
    const float* xr = x + ((size_t)b * N_ + n)  * C_;
    const float* fr = x + ((size_t)b * N_ + id) * C_;
    __builtin_prefetch(fr, 0, 0);            // global_prefetch_b8 (gathered row)
    for (int c = 0; c < C_; c += 4) {
      const float4 xa = *(const float4*)(xr + c);
      const float4 fa = *(const float4*)(fr + c);
      edgeT[tid][c + 0] = (_Float16)(fa.x - xa.x);
      edgeT[tid][c + 1] = (_Float16)(fa.y - xa.y);
      edgeT[tid][c + 2] = (_Float16)(fa.z - xa.z);
      edgeT[tid][c + 3] = (_Float16)(fa.w - xa.w);
      edgeT[tid][C_ + c + 0] = (_Float16)xa.x;
      edgeT[tid][C_ + c + 1] = (_Float16)xa.y;
      edgeT[tid][C_ + c + 2] = (_Float16)xa.z;
      edgeT[tid][C_ + c + 3] = (_Float16)xa.w;
    }
  }

  // ---- weight fragments in contiguous-k B layout (k = 16h + s): wave w builds coltile ct=w ------
  {
    const int col  = w * 16 + (lane & 15);   // output channel
    const int koff = (lane >> 4) * 16;       // K half split at lane 16
    for (int kt = 0; kt < 2; ++kt) {
      for (int j = 0; j < 8; ++j) {
        const int Kk = kt * 32 + koff + 2 * j;
        const float lo = (Kk     < 48) ? w1[col * 48 + Kk]     : 0.0f;
        const float hi = (Kk + 1 < 48) ? w1[col * 48 + Kk + 1] : 0.0f;
        B1f[w][kt][lane][j] = pkh(lo, hi);
        B2f[w][kt][lane][j] = pkh(w2[col * 64 + Kk], w2[col * 64 + Kk + 1]);
      }
    }
  }
  __syncthreads();

  const int khalf = (lane >> 4) * 16;        // A layout: K half split at lane 16
  float nmax[4] = {-1e30f, -1e30f, -1e30f, -1e30f};

  for (int t = 0; t < 2; ++t) {              // two 16-row tiles = one point
    const int tilebase = w * 32 + t * 16;
    const int arow     = tilebase + (lane & 15);

    // A1 fragments from LDS edge tile (K = 48, padded to 64)
    Frag16 A1[2];
    for (int kt = 0; kt < 2; ++kt) {
      for (int j = 0; j < 8; ++j) {
        const int c = kt * 32 + khalf + 2 * j;
        A1[kt].u[j] = (c < 48) ? *(const uint32_t*)&edgeT[arow][c] : 0u;
      }
    }

    // ---- GEMM1 + epilogue --------------------------------------------------
    for (int ct = 0; ct < 4; ++ct) {
      v8f acc = {};
      for (int kt = 0; kt < 2; ++kt) {
        Frag16 Bb;
        for (int j = 0; j < 8; ++j) Bb.u[j] = B1f[ct][kt][lane][j];
        acc = wmma16(A1[kt], Bb, acc);
      }
      if (MODE == 0) {
        float s = 0.0f, q = 0.0f;
        for (int r = 0; r < 8; ++r) { const float v = acc[r]; s += v; q += v * v; }
        s += __shfl_xor(s, 16);
        q += __shfl_xor(q, 16);
        if (lane < 16) {
          atomicAdd(&ssum[ct * 16 + lane], s);   // ds_add_f32
          atomicAdd(&ssq [ct * 16 + lane], q);
        }
        continue;
      }
      const int   c   = ct * 16 + (lane & 15);
      const float a1c = ws[256 + c], b1c = ws[320 + c];
      for (int r = 0; r < 8; ++r) {
        float v = acc[r] * a1c + b1c;
        v = (v >= 0.0f) ? v : SLOPE_ * v;
        h1T[w][r + 8 * (lane >> 4)][c] = (_Float16)v;   // C->A transpose via LDS
      }
    }
    if (MODE == 0) continue;

    // A2 fragments from per-wave h1 tile (same-wave LDS ops are in-order)
    Frag16 A2[2];
    for (int kt = 0; kt < 2; ++kt) {
      for (int j = 0; j < 8; ++j) {
        const int c = kt * 32 + khalf + 2 * j;
        A2[kt].u[j] = *(const uint32_t*)&h1T[w][lane & 15][c];
      }
    }

    // ---- GEMM2 + epilogue --------------------------------------------------
    for (int ct = 0; ct < 4; ++ct) {
      v8f acc2 = {};
      for (int kt = 0; kt < 2; ++kt) {
        Frag16 Bb;
        for (int j = 0; j < 8; ++j) Bb.u[j] = B2f[ct][kt][lane][j];
        acc2 = wmma16(A2[kt], Bb, acc2);
      }
      if (MODE == 1) {
        float s = 0.0f, q = 0.0f;
        for (int r = 0; r < 8; ++r) { const float v = acc2[r]; s += v; q += v * v; }
        s += __shfl_xor(s, 16);
        q += __shfl_xor(q, 16);
        if (lane < 16) {
          atomicAdd(&ssum[ct * 16 + lane], s);
          atomicAdd(&ssq [ct * 16 + lane], q);
        }
      } else if (MODE == 2) {
        const int   c   = ct * 16 + (lane & 15);
        const float a2c = ws[384 + c], b2c = ws[448 + c];
        float mx = -1e30f;
        for (int r = 0; r < 8; ++r) {
          float v = acc2[r] * a2c + b2c;
          v = (v >= 0.0f) ? v : SLOPE_ * v;
          mx = fmaxf(mx, v);
        }
        mx = fmaxf(mx, __shfl_xor(mx, 16));   // combine row halves (lanes l, l+16)
        nmax[ct] = fmaxf(nmax[ct], mx);       // combine the two k-tiles
      }
    }
  }

  if (MODE == 2) {
    if (lane < 16) {
      const int p = wg * 4 + w;               // output point index (b*N + n)
      for (int ct = 0; ct < 4; ++ct)
        out[(size_t)p * O_ + ct * 16 + lane] = nmax[ct];
    }
  } else {
    __syncthreads();
    float* gsum = ws + (MODE == 0 ? 0 : 128);
    if (tid < 64) {
      atomicAdd(&gsum[tid],      ssum[tid]);
      atomicAdd(&gsum[64 + tid], ssq[tid]);
    }
  }
}

// ---------------------------------------------------------------------------
__global__ void finalize_kernel(float* __restrict__ ws,
                                const float* __restrict__ gamma,
                                const float* __restrict__ beta,
                                int stage) {
  const int t = threadIdx.x;
  if (t >= 64) return;
  const float  Mf   = (float)M_TOT;
  const float* sums = ws + stage * 128;
  const float  mean = sums[t] / Mf;
  const float  var  = sums[64 + t] / Mf - mean * mean;
  const float  rstd = rsqrtf(var + EPS_);
  const float  a    = rstd * gamma[t];
  const float  b    = beta[t] - mean * a;
  ws[256 + stage * 128 + t] = a;
  ws[320 + stage * 128 + t] = b;
}

// ---------------------------------------------------------------------------
extern "C" void kernel_launch(void* const* d_in, const int* in_sizes, int n_in,
                              void* d_out, int out_size, void* d_ws, size_t ws_size,
                              hipStream_t stream) {
  const float* x   = (const float*)d_in[0];
  const float* pos = (const float*)d_in[1];
  const float* w1  = (const float*)d_in[2];
  const float* g1  = (const float*)d_in[3];
  const float* b1  = (const float*)d_in[4];
  const float* w2  = (const float*)d_in[5];
  const float* g2  = (const float*)d_in[6];
  const float* b2  = (const float*)d_in[7];

  float* outp   = (float*)d_out;                         // (B,N,O) f32
  int*   idxout = (int*)d_out + (size_t)B_ * N_ * O_;    // (B,N,K) i32 bits
  float* wsf    = (float*)d_ws;

  init_ws_kernel<<<1, 256, 0, stream>>>(wsf);

  const size_t knn_lds = (size_t)3 * N_ * sizeof(float) + 8 * 33 * sizeof(uint64_t);
  knn_kernel<<<(B_ * N_) / 8, 256, knn_lds, stream>>>(pos, idxout);

  const int fgrid = M_TOT / 128;   // 4096
  fused_kernel<0><<<fgrid, 128, 0, stream>>>(x, idxout, w1, w2, wsf, outp);
  finalize_kernel<<<1, 64, 0, stream>>>(wsf, g1, b1, 0);
  fused_kernel<1><<<fgrid, 128, 0, stream>>>(x, idxout, w1, w2, wsf, outp);
  finalize_kernel<<<1, 64, 0, stream>>>(wsf, g2, b2, 1);
  fused_kernel<2><<<fgrid, 128, 0, stream>>>(x, idxout, w1, w2, wsf, outp);
}
